// LSTM_19318762897922
// MI455X (gfx1250) — matrix operand order repaired and checked by reading, not compiled, not run on hardware
//
#include <hip/hip_runtime.h>

typedef _Float16 v16h __attribute__((ext_vector_type(16)));
typedef float    v8f  __attribute__((ext_vector_type(8)));
typedef unsigned int u32x4 __attribute__((ext_vector_type(4)));
typedef int          i32x8 __attribute__((ext_vector_type(8)));
typedef int          i32x4 __attribute__((ext_vector_type(4)));

#define T_STEPS 2048
#define BATCH   64
#define INP     512
#define HID     512
#define KDIM    1024
#define FOURH   2048
#define NWG     64
#define BH      (BATCH * HID)
#define LPAD    8          // 8-half (16B) row pad: 1040B stride -> banks fan out

#if defined(__has_builtin)
#if __has_builtin(__builtin_amdgcn_tensor_load_to_lds) && \
    __has_builtin(__builtin_amdgcn_s_wait_tensorcnt)
#define HAVE_TDM 1
#endif
#endif
#ifndef HAVE_TDM
#define HAVE_TDM 0
#endif

// ---------------------------------------------------------------------------
// Prep: gate-concatenated transposed fp16 weights WT[4H][K]; zero h ping-pong
// buffers and the grid-barrier counter.
// ---------------------------------------------------------------------------
__global__ void lstm_prep(const float* __restrict__ Wxf, const float* __restrict__ Wxi,
                          const float* __restrict__ Wxo, const float* __restrict__ Wxc,
                          const float* __restrict__ Whf, const float* __restrict__ Whi,
                          const float* __restrict__ Who, const float* __restrict__ Whc,
                          _Float16* __restrict__ WT, _Float16* __restrict__ h16,
                          unsigned* __restrict__ bar)
{
    size_t idx = (size_t)blockIdx.x * blockDim.x + threadIdx.x;
    if (idx < (size_t)FOURH * KDIM) {
        int n    = (int)(idx >> 10);
        int k    = (int)(idx & 1023);
        int gate = n >> 9;
        int col  = n & 511;
        const float* Wx = (gate == 0) ? Wxf : (gate == 1) ? Wxi : (gate == 2) ? Wxo : Wxc;
        const float* Wh = (gate == 0) ? Whf : (gate == 1) ? Whi : (gate == 2) ? Who : Whc;
        float v = (k < INP) ? Wx[(size_t)k * HID + col]
                            : Wh[(size_t)(k - INP) * HID + col];
        WT[idx] = (_Float16)v;
    }
    if (idx < (size_t)2 * BH) h16[idx] = (_Float16)0.0f;
    if (idx == 0) *bar = 0u;
}

// ---- helpers --------------------------------------------------------------
__device__ __forceinline__ void load_x8(float4 px[8], const float* base)
{
    const float4* p = (const float4*)base;
#pragma unroll
    for (int i = 0; i < 8; ++i) px[i] = p[i];
}

__device__ __forceinline__ void cvt_store32(const float4 px[8], _Float16* dst)
{
    union { _Float16 h[32]; uint4 q[4]; } u;
    const float* f = (const float*)px;
#pragma unroll
    for (int i = 0; i < 32; ++i) u.h[i] = (_Float16)f[i];
    uint4* d = (uint4*)dst;
#pragma unroll
    for (int j = 0; j < 4; ++j) d[j] = u.q[j];
}

// ---------------------------------------------------------------------------
// Persistent LSTM: 64 WGs x 256 threads (8 wave32). Per step each wave does a
// 16x16 WMMA tile over K=1024; A-tile staged in LDS (x: register-prefetched
// one step ahead + converted; h: TDM tensor_load_to_lds with LDS padding).
// ---------------------------------------------------------------------------
__global__ void __launch_bounds__(256, 1)
lstm_persistent(const float* __restrict__ X,
                const float* __restrict__ bf, const float* __restrict__ bi,
                const float* __restrict__ bo, const float* __restrict__ bc,
                const _Float16* __restrict__ WT,
                _Float16* __restrict__ h16,
                unsigned* __restrict__ bar,
                float* __restrict__ out)
{
    const int tid   = threadIdx.x;
    const int wave  = tid >> 5;
    const int lane  = tid & 31;
    const int wg    = blockIdx.x;
    const int m0    = (wg >> 4) * 16;     // batch-row tile base
    const int npair = wg & 15;            // 32-wide H-column block
    const int gate  = wave & 3;
    const int nhalf = wave >> 2;
    const int g     = lane >> 4;
    const int l15   = lane & 15;

    const float* bptr = (gate == 0) ? bf : (gate == 1) ? bi : (gate == 2) ? bo : bc;
    const float  bias = bptr[npair * 32 + nhalf * 16 + l15];

    const _Float16* wrow =
        WT + ((size_t)(gate * HID + npair * 32 + nhalf * 16 + l15)) * KDIM;

    __shared__ _Float16 xbuf[2][16][INP + LPAD];   // ping-pong x tile (fp16)
    __shared__ _Float16 hbuf[16][HID + LPAD];      // h tile (fp16)
    __shared__ float    gv[4][16][32];

    // staging mapping: thread covers row xr, 32 consecutive columns at xc
    const int xr = tid >> 4;              // 0..15
    const int xc = (tid & 15) * 32;       // 0..480

    // elementwise mapping
    const int col_e = tid & 31;
    const int rowA  = tid >> 5;
    const int rowB  = rowA + 8;
    const int bA    = m0 + rowA;
    const int bB    = m0 + rowB;
    const int hcA   = npair * 32 + col_e;
    float cA = 0.0f, cB = 0.0f;

#if HAVE_TDM
    // TDM descriptor pieces that never change (16 x 512-half tile, 2B elems,
    // pad 4 dwords every 256 dwords => LDS row stride 520 halfs)
    const unsigned lds_byte = (unsigned)(unsigned long long)(size_t)&hbuf[0][0];
    i32x8 g1;
    g1[0] = (int)((1u << 16) | (1u << 20) | (7u << 22) | (3u << 25));
    g1[1] = (int)(512u << 16);   // tensor_dim0[15:0]=512 in bits[63:48]
    g1[2] = (int)(16u << 16);    // tensor_dim1[15:0]=16 in bits[95:80]
    g1[3] = (int)(512u << 16);   // tile_dim0=512 in bits[127:112]
    g1[4] = (int)16;             // tile_dim1=16
    g1[5] = (int)512;            // tensor_dim0_stride=512
    g1[6] = 0; g1[7] = 0;
    const i32x4 gz4 = {0, 0, 0, 0};
    const i32x8 gz8 = {0, 0, 0, 0, 0, 0, 0, 0};
#endif

    // ---- prologue: stage x(0) into xbuf[0]
    {
        float4 px[8];
        load_x8(px, X + ((size_t)0 * BATCH + m0 + xr) * INP + xc);
        cvt_store32(px, &xbuf[0][xr][xc]);
    }

    for (int t = 0; t < T_STEPS; ++t) {
        // ---- 1. issue x(t+1) prefetch loads (overlaps the whole K-loop)
        const int tn = (t + 1 < T_STEPS) ? (t + 1) : t;
        float4 px[8];
        load_x8(px, X + ((size_t)tn * BATCH + m0 + xr) * INP + xc);

        // ---- 2. stage h(t) tile into LDS
#if HAVE_TDM
        if (wave == 0) {
            unsigned long long ga = (unsigned long long)(size_t)
                (h16 + (size_t)(t & 1) * BH + (size_t)m0 * HID);
            u32x4 g0;
            g0.x = 1u;                                    // count=1 (valid user D#)
            g0.y = lds_byte;                              // lds_addr
            g0.z = (unsigned)(ga & 0xFFFFFFFFu);          // global_addr[31:0]
            g0.w = (unsigned)((ga >> 32) & 0x01FFFFFFu)   // global_addr[56:32]
                 | (2u << 30);                            // type=2 (image)
            __builtin_amdgcn_tensor_load_to_lds(g0, g1, gz4, gz4, gz8, 0);
            __builtin_amdgcn_s_wait_tensorcnt(0);
        }
#else
        {
            const uint4* hs = (const uint4*)(h16 + (size_t)(t & 1) * BH
                                             + (size_t)(m0 + xr) * HID + xc);
            uint4* hd = (uint4*)&hbuf[xr][xc];
#pragma unroll
            for (int j = 0; j < 4; ++j) hd[j] = hs[j];
        }
#endif
        __syncthreads();

        // ---- 3. K-loop: 32 x v_wmma_f32_16x16x32_f16, A from LDS, B hoisted
        v8f acc = {0.0f, 0.0f, 0.0f, 0.0f, 0.0f, 0.0f, 0.0f, 0.0f};
        const _Float16* arow_x = &xbuf[t & 1][l15][0];
        const _Float16* arow_h = &hbuf[l15][0];

#pragma unroll
        for (int kk = 0; kk < INP; kk += 32) {
            v16h a;
            ((uint4*)&a)[0] = *(const uint4*)(arow_x + kk + g * 8);
            ((uint4*)&a)[1] = *(const uint4*)(arow_x + kk + 16 + g * 8);
            v16h b;
            ((uint4*)&b)[0] = *(const uint4*)(wrow + kk + g * 8);
            ((uint4*)&b)[1] = *(const uint4*)(wrow + kk + 16 + g * 8);
            acc = __builtin_amdgcn_wmma_f32_16x16x32_f16(
                      false, a, false, b, (short)0, acc, false, false);
        }
#pragma unroll
        for (int kk = 0; kk < HID; kk += 32) {
            v16h a;
            ((uint4*)&a)[0] = *(const uint4*)(arow_h + kk + g * 8);
            ((uint4*)&a)[1] = *(const uint4*)(arow_h + kk + 16 + g * 8);
            v16h b;
            ((uint4*)&b)[0] = *(const uint4*)(wrow + INP + kk + g * 8);
            ((uint4*)&b)[1] = *(const uint4*)(wrow + INP + kk + 16 + g * 8);
            acc = __builtin_amdgcn_wmma_f32_16x16x32_f16(
                      false, a, false, b, (short)0, acc, false, false);
        }

        // ---- 4. retire x(t+1) prefetch into the other LDS buffer
        cvt_store32(px, &xbuf[(t + 1) & 1][xr][xc]);

        // ---- 5. bias + activation -> LDS gate tiles
#pragma unroll
        for (int r = 0; r < 8; ++r) {
            float v = acc[r] + bias;
            v = (gate < 3) ? (1.0f / (1.0f + __expf(-v))) : tanhf(v);
            gv[gate][g * 8 + r][nhalf * 16 + l15] = v;
        }
        __syncthreads();

        // ---- 6. elementwise cell update (c in registers)
        {
            float fA = gv[0][rowA][col_e], iA = gv[1][rowA][col_e];
            float oA = gv[2][rowA][col_e], gA = gv[3][rowA][col_e];
            float fB = gv[0][rowB][col_e], iB = gv[1][rowB][col_e];
            float oB = gv[2][rowB][col_e], gB = gv[3][rowB][col_e];
            cA = fA * cA + iA * gA;
            cB = fB * cB + iB * gB;
            float hA = oA * tanhf(cA);
            float hB = oB * tanhf(cB);

            out[((size_t)t * BATCH + bA) * HID + hcA] = hA;
            out[((size_t)t * BATCH + bB) * HID + hcA] = hB;

            _Float16* hnext = h16 + (size_t)((t + 1) & 1) * BH;
            hnext[(size_t)bA * HID + hcA] = (_Float16)hA;
            hnext[(size_t)bB * HID + hcA] = (_Float16)hB;

            if (t == T_STEPS - 1) {
                size_t hT = (size_t)T_STEPS * BH;
                size_t cT = hT + (size_t)BH;
                out[hT + (size_t)bA * HID + hcA] = hA;
                out[hT + (size_t)bB * HID + hcA] = hB;
                out[cT + (size_t)bA * HID + hcA] = cA;
                out[cT + (size_t)bB * HID + hcA] = cB;
            }
        }
        __syncthreads();

        // ---- 7. grid-wide barrier (monotonic arrival counter)
        __threadfence();
        if (tid == 0) {
            __hip_atomic_fetch_add(bar, 1u, __ATOMIC_RELEASE,
                                   __HIP_MEMORY_SCOPE_AGENT);
            const unsigned target = (unsigned)NWG * (unsigned)(t + 1);
            while (__hip_atomic_load(bar, __ATOMIC_ACQUIRE,
                                     __HIP_MEMORY_SCOPE_AGENT) < target) {
                __builtin_amdgcn_s_sleep(2);
            }
        }
        __syncthreads();
    }
}

// ---------------------------------------------------------------------------
extern "C" void kernel_launch(void* const* d_in, const int* in_sizes, int n_in,
                              void* d_out, int out_size, void* d_ws, size_t ws_size,
                              hipStream_t stream)
{
    (void)in_sizes; (void)n_in; (void)out_size; (void)ws_size;

    const float* X   = (const float*)d_in[0];
    const float* Wxf = (const float*)d_in[1];
    const float* Whf = (const float*)d_in[2];
    const float* bf  = (const float*)d_in[3];
    const float* Wxi = (const float*)d_in[4];
    const float* Whi = (const float*)d_in[5];
    const float* bi  = (const float*)d_in[6];
    const float* Wxo = (const float*)d_in[7];
    const float* Who = (const float*)d_in[8];
    const float* bo  = (const float*)d_in[9];
    const float* Wxc = (const float*)d_in[10];
    const float* Whc = (const float*)d_in[11];
    const float* bc  = (const float*)d_in[12];

    _Float16* WT  = (_Float16*)d_ws;                     // 4H x K fp16 (4 MB)
    _Float16* h16 = WT + (size_t)FOURH * KDIM;           // 2 x B x H fp16
    unsigned* bar = (unsigned*)(h16 + (size_t)2 * BH);
    float*    out = (float*)d_out;

    const int prep_elems = FOURH * KDIM;
    lstm_prep<<<(prep_elems + 255) / 256, 256, 0, stream>>>(
        Wxf, Wxi, Wxo, Wxc, Whf, Whi, Who, Whc, WT, h16, bar);

    lstm_persistent<<<NWG, 256, 0, stream>>>(
        X, bf, bi, bo, bc, WT, h16, bar, out);
}
